// SCNystromformerSelfAttention_39668317946182
// MI455X (gfx1250) — matrix operand order repaired and checked by reading, not compiled, not run on hardware
//
#include <hip/hip_runtime.h>
#include <hip/hip_bf16.h>

// ---------------------------------------------------------------------------
// Nystromformer self-attention for MI455X (gfx1250, wave32, WMMA).
// - All GEMMs (incl. the depthwise conv, expressed as a banded matmul) use
//   v_wmma_f32_16x16x32_f16 (f16 in, f32 accumulate).
// - All global->LDS tile staging uses global_load_async_to_lds_b128
//   (ASYNCcnt) -- inputs are pre-converted/pre-transposed so every staging
//   copy is a pure strided-row b128 copy.
// - LDS rows padded (72 or 104 halves) so fragment b128 loads are
//   bank-conflict-free.
// ---------------------------------------------------------------------------

typedef __attribute__((ext_vector_type(16))) _Float16 v16h;
typedef __attribute__((ext_vector_type(8)))  _Float16 v8h;
typedef __attribute__((ext_vector_type(4)))  _Float16 v4h;
typedef __attribute__((ext_vector_type(8)))  float    v8f;
typedef __attribute__((ext_vector_type(4)))  float    v4f;

#define BB  4
#define SS  4096
#define EE  1024
#define HH  16
#define DD  64
#define LL  64
#define KSZ 33
#define LP  72    // padded LDS row stride (halves) for K=64 tiles
#define LPW 104   // padded LDS row stride (halves) for K=96 tiles

// ---------------------------------------------------------------------------
// Async global->LDS staging (GLOBAL_LOAD_ASYNC_TO_LDS_B128, ASYNCcnt).
// Generic LDS pointer low 32 bits == DS byte offset on AMDGPU.
// ---------------------------------------------------------------------------
__device__ __forceinline__ void async_chunk(_Float16* dst, const _Float16* src) {
  unsigned lds = (unsigned)(size_t)dst;
  asm volatile("global_load_async_to_lds_b128 %0, %1, off"
               :: "v"(lds), "v"(src) : "memory");
}
__device__ __forceinline__ void async_wait() {
  asm volatile("s_wait_asynccnt 0x0" ::: "memory");
}
// rows x 64-half tile, 8 chunks/row.
__device__ __forceinline__ void async_tile8(_Float16* dst, int dstStride,
                                            const _Float16* src, size_t srcStride,
                                            int rows, int tid) {
  for (int c = tid; c < rows * 8; c += 128) {
    int r = c >> 3, c0 = (c & 7) * 8;
    async_chunk(dst + r * dstStride + c0, src + (size_t)r * srcStride + c0);
  }
}
// 64 x 96-half tile, 12 chunks/row.
__device__ __forceinline__ void async_tile12(_Float16* dst, int dstStride,
                                             const _Float16* src, size_t srcStride,
                                             int tid) {
  for (int c = tid; c < 64 * 12; c += 128) {
    int r = c / 12, c0 = (c % 12) * 8;
    async_chunk(dst + r * dstStride + c0, src + (size_t)r * srcStride + c0);
  }
}

// ---------------------------------------------------------------------------
// WMMA fragment helpers (wave32 layouts per CDNA5 ISA 7.12.2).
//  A row-major [m][k]; B stored TRANSPOSED [n][k] so lane data is contiguous.
// ---------------------------------------------------------------------------
__device__ __forceinline__ v16h ld16(const _Float16* p0, const _Float16* p1) {
  v8h a = *(const v8h*)p0;
  v8h b = *(const v8h*)p1;
  return __builtin_shufflevector(a, b, 0, 1, 2, 3, 4, 5, 6, 7,
                                 8, 9, 10, 11, 12, 13, 14, 15);
}
__device__ __forceinline__ v16h load_a_frag(const _Float16* A, int lda, int m,
                                            int k0, int lane) {
  const _Float16* p = A + m * lda + k0 + ((lane & 16) ? 8 : 0);
  return ld16(p, p + 16);
}
__device__ __forceinline__ v16h load_bT_frag(const _Float16* BT, int ldb, int n,
                                             int k0, int lane) {
  const _Float16* p = BT + n * ldb + k0 + ((lane & 16) ? 8 : 0);
  return ld16(p, p + 16);
}

// 64x64xKT tile per 128-thread block (4 waves; wave w rows 16w..16w+15).
template <int KT>
__device__ __forceinline__ void wmma_accum_g(v8f acc[4], const _Float16* A, int lda,
                                             const _Float16* BT, int ldb,
                                             int wave, int lane) {
  const int m  = wave * 16 + (lane & 15);
  const int nl = lane & 15;
#pragma unroll
  for (int k0 = 0; k0 < KT; k0 += 32) {
    v16h a = load_a_frag(A, lda, m, k0, lane);
#pragma unroll
    for (int j = 0; j < 4; ++j) {
      v16h b = load_bT_frag(BT, ldb, j * 16 + nl, k0, lane);
      acc[j] = __builtin_amdgcn_wmma_f32_16x16x32_f16(
          false, a, false, b, (short)0, acc[j], false, false);
    }
  }
}

// C/D 16x16 f32: vgpr i -> M=i+8*(lane/16), N=lane%16.
__device__ __forceinline__ void store_acc_lds(const v8f acc[4], float* C, int ldc,
                                              int wave, int lane) {
  const int row0 = wave * 16 + ((lane & 16) ? 8 : 0);
  const int col  = lane & 15;
#pragma unroll
  for (int j = 0; j < 4; ++j)
#pragma unroll
    for (int i = 0; i < 8; ++i)
      C[(row0 + i) * ldc + j * 16 + col] = acc[j][i];
}

// ---------------------------------------------------------------------------
// Prep kernels: hidden f32->f16; W f32 -> W^T f16 (LDS-tiled); v -> v^T f16.
// ---------------------------------------------------------------------------
__global__ void convert_hidden_kernel(const float* __restrict__ src,
                                      _Float16* __restrict__ dst) {
  size_t i = ((size_t)blockIdx.x * 256 + threadIdx.x) * 4;
  v4f x = *(const v4f*)(src + i);
  v4h h;
  h[0] = (_Float16)x[0]; h[1] = (_Float16)x[1];
  h[2] = (_Float16)x[2]; h[3] = (_Float16)x[3];
  *(v4h*)(dst + i) = h;
}

__global__ void __launch_bounds__(128)
transposeW_kernel(const float* __restrict__ W, _Float16* __restrict__ WT) {
  __shared__ _Float16 t[64 * LP];
  const int tid = threadIdx.x;
  const int k0 = blockIdx.x * 64, n0 = blockIdx.y * 64;
  for (int idx = tid; idx < 64 * 64; idx += 128) {
    int k = idx >> 6, n = idx & 63;
    t[n * LP + k] = (_Float16)W[(size_t)(k0 + k) * EE + n0 + n];
  }
  __syncthreads();
  for (int c = tid; c < 64 * 8; c += 128) {
    int n = c >> 3, c0 = (c & 7) * 8;
    *(v8h*)(WT + (size_t)(n0 + n) * EE + k0 + c0) = *(const v8h*)(t + n * LP + c0);
  }
}

__global__ void __launch_bounds__(128)
transposeV_kernel(const _Float16* __restrict__ v, _Float16* __restrict__ vT) {
  const int bh = blockIdx.x, s0 = blockIdx.y * 64;
  for (int idx = threadIdx.x; idx < 64 * 64; idx += 128) {
    int d = idx >> 6, s = idx & 63;
    vT[((size_t)bh * DD + d) * SS + s0 + s] = v[((size_t)bh * SS + s0 + s) * DD + d];
  }
}

__global__ void zero_u32_kernel(unsigned* p) { *p = 0u; }

// ---------------------------------------------------------------------------
// K1: QKV projection GEMM, 128x64 tile per block; each wave owns 32 rows so
// every B fragment feeds two WMMAs. All staging async. grid=(B*S/128, H).
// ---------------------------------------------------------------------------
__global__ void __launch_bounds__(128)
qkv_gemm_kernel(const _Float16* __restrict__ X16, const _Float16* __restrict__ WT,
                const float* __restrict__ bias, float scale,
                _Float16* __restrict__ out) {
  __shared__ _Float16 As[128 * LP];
  __shared__ _Float16 BT[64 * LP];
  __shared__ _Float16 Cs[128 * 64];
  __shared__ float biasS[64];
  const int tid = threadIdx.x, wave = tid >> 5, lane = tid & 31;
  const int row0 = blockIdx.x * 128;
  const int h = blockIdx.y, col0 = h * 64;
  if (tid < 64) biasS[tid] = bias[col0 + tid];
  v8f acc[2][4] = {};
  for (int kc = 0; kc < EE; kc += 64) {
    async_tile8(As, LP, X16 + (size_t)row0 * EE + kc, EE, 128, tid);
    async_tile8(BT, LP, WT + (size_t)col0 * EE + kc, EE, 64, tid);
    async_wait();
    __syncthreads();
    const int m0 = wave * 32, nl = lane & 15;
#pragma unroll
    for (int k0 = 0; k0 < 64; k0 += 32) {
      v16h a0 = load_a_frag(As, LP, m0 + nl, k0, lane);
      v16h a1 = load_a_frag(As, LP, m0 + 16 + nl, k0, lane);
#pragma unroll
      for (int j = 0; j < 4; ++j) {
        v16h b = load_bT_frag(BT, LP, j * 16 + nl, k0, lane);
        acc[0][j] = __builtin_amdgcn_wmma_f32_16x16x32_f16(
            false, a0, false, b, (short)0, acc[0][j], false, false);
        acc[1][j] = __builtin_amdgcn_wmma_f32_16x16x32_f16(
            false, a1, false, b, (short)0, acc[1][j], false, false);
      }
    }
    __syncthreads();
  }
#pragma unroll
  for (int t = 0; t < 2; ++t)
#pragma unroll
    for (int j = 0; j < 4; ++j)
#pragma unroll
      for (int i = 0; i < 8; ++i) {
        int row = wave * 32 + t * 16 + i + ((lane & 16) ? 8 : 0);
        int col = j * 16 + (lane & 15);
        Cs[row * 64 + col] = (_Float16)((acc[t][j][i] + biasS[col]) * scale);
      }
  __syncthreads();
  for (int c = tid; c < 128 * 8; c += 128) {
    int r = c >> 3, c0 = (c & 7) * 8;
    int row = row0 + r, b = row >> 12, s = row & (SS - 1);
    *(v8h*)(out + (((size_t)b * HH + h) * SS + s) * DD + c0) =
        *(const v8h*)(Cs + r * 64 + c0);
  }
}

// ---------------------------------------------------------------------------
// K2: landmark means. grid = B*H*L blocks of 64 threads (one per d).
// ---------------------------------------------------------------------------
__global__ void __launch_bounds__(64)
landmark_kernel(const _Float16* __restrict__ q, const _Float16* __restrict__ k,
                _Float16* __restrict__ ql, _Float16* __restrict__ kl) {
  const int l  = blockIdx.x & (LL - 1);
  const int bh = blockIdx.x >> 6;
  const int d  = threadIdx.x;
  const size_t base = ((size_t)bh * SS + (size_t)l * (SS / LL)) * DD + d;
  float sq = 0.f, sk = 0.f;
  for (int i = 0; i < SS / LL; ++i) {
    sq += (float)q[base + (size_t)i * DD];
    sk += (float)k[base + (size_t)i * DD];
  }
  const size_t o = ((size_t)bh * LL + l) * DD + d;
  ql[o] = (_Float16)(sq * (1.f / (SS / LL)));
  kl[o] = (_Float16)(sk * (1.f / (SS / LL)));
}

// ---------------------------------------------------------------------------
// K3: kernel_2 = softmax(ql @ kl^T) per (b,h) + global max of column sums
// via float-bit atomicMax (positive floats -> bit order == value order).
// ---------------------------------------------------------------------------
__global__ void __launch_bounds__(128)
kernel2_kernel(const _Float16* __restrict__ ql, const _Float16* __restrict__ kl,
               float* __restrict__ k2, unsigned* __restrict__ gmax) {
  __shared__ _Float16 As[64 * LP], BT[64 * LP];
  __shared__ float Ls[64 * 64];
  __shared__ float colsum[64];
  const int bh = blockIdx.x, tid = threadIdx.x, wave = tid >> 5, lane = tid & 31;
  async_tile8(As, LP, ql + (size_t)bh * 4096, 64, 64, tid);
  async_tile8(BT, LP, kl + (size_t)bh * 4096, 64, 64, tid);  // (kl^T)^T = kl
  async_wait();
  __syncthreads();
  v8f acc[4] = {};
  wmma_accum_g<64>(acc, As, LP, BT, LP, wave, lane);
  store_acc_lds(acc, Ls, 64, wave, lane);
  __syncthreads();
  if (tid < 64) {
    float mx = -3.0e38f;
    for (int c = 0; c < 64; ++c) mx = fmaxf(mx, Ls[tid * 64 + c]);
    float s = 0.f;
    for (int c = 0; c < 64; ++c) {
      float e = __expf(Ls[tid * 64 + c] - mx);
      Ls[tid * 64 + c] = e; s += e;
    }
    float inv = 1.f / s;
    for (int c = 0; c < 64; ++c) {
      float p = Ls[tid * 64 + c] * inv;
      Ls[tid * 64 + c] = p;
      k2[(size_t)bh * 4096 + tid * 64 + c] = p;
    }
  }
  __syncthreads();
  if (tid < 64) {
    float cs = 0.f;
    for (int r = 0; r < 64; ++r) cs += Ls[r * 64 + tid];
    colsum[tid] = cs;
  }
  __syncthreads();
  if (tid == 0) {
    float mx = 0.f;
    for (int c = 0; c < 64; ++c) mx = fmaxf(mx, colsum[c]);
    atomicMax(gmax, __float_as_uint(mx));
  }
}

// ---------------------------------------------------------------------------
// K4: iterative pinv (6 Newton iterations, 64x64 f32 in LDS; f32 VALU for
// precision). Output stored TRANSPOSED (pinvT[l2][l1]) for async B staging.
// ---------------------------------------------------------------------------
__device__ __forceinline__ void mm64f(float* C, const float* A, const float* B,
                                      float diag, float scale, int tid) {
  const int row = tid >> 2, c0 = (tid & 3) * 16;
  float r[16];
#pragma unroll
  for (int i = 0; i < 16; ++i) r[i] = 0.f;
  for (int k = 0; k < 64; ++k) {
    float a = A[row * 64 + k];
    const float* Bp = B + k * 64 + c0;
#pragma unroll
    for (int i = 0; i < 16; ++i) r[i] = fmaf(a, Bp[i], r[i]);
  }
  __syncthreads();
#pragma unroll
  for (int i = 0; i < 16; ++i) {
    int c = c0 + i;
    C[row * 64 + c] = scale * r[i] + ((c == row) ? diag : 0.f);
  }
  __syncthreads();
}

__device__ __forceinline__ void mm64h(float* C, const _Float16* A, const float* B,
                                      int tid) {
  const int row = tid >> 2, c0 = (tid & 3) * 16;
  float r[16];
#pragma unroll
  for (int i = 0; i < 16; ++i) r[i] = 0.f;
  for (int k = 0; k < 64; ++k) {
    float a = (float)A[row * 64 + k];
    const float* Bp = B + k * 64 + c0;
#pragma unroll
    for (int i = 0; i < 16; ++i) r[i] = fmaf(a, Bp[i], r[i]);
  }
  __syncthreads();
#pragma unroll
  for (int i = 0; i < 16; ++i) C[row * 64 + c0 + i] = r[i];
  __syncthreads();
}

__global__ void __launch_bounds__(256)
pinv_kernel(const float* __restrict__ k2, const unsigned* __restrict__ gmax,
            _Float16* __restrict__ pinvT) {
  __shared__ _Float16 A16[4096];
  __shared__ float V[4096], KV[4096], T[4096];
  const int bh = blockIdx.x, tid = threadIdx.x;
  const float inv = 1.f / __uint_as_float(*gmax);
  for (int i = tid; i < 4096; i += 256) A16[i] = (_Float16)k2[(size_t)bh * 4096 + i];
  for (int i = tid; i < 4096; i += 256) {
    int r = i >> 6, c = i & 63;
    V[i] = k2[(size_t)bh * 4096 + c * 64 + r] * inv;  // a^T / gmax
  }
  __syncthreads();
  for (int it = 0; it < 6; ++it) {
    mm64h(KV, A16, V, tid);  // KV = a @ V
    for (int i = tid; i < 4096; i += 256) {
      int r = i >> 6, c = i & 63;
      T[i] = ((r == c) ? 7.f : 0.f) - KV[i];  // T = 7I - KV
    }
    __syncthreads();
    mm64f(T, KV, T, 15.f, -1.f, tid);  // T = 15I - KV@T
    mm64f(T, KV, T, 13.f, -1.f, tid);  // T = 13I - KV@T
    mm64f(V, V, T, 0.f, 0.25f, tid);   // V = 0.25 * V@T
  }
  for (int i = tid; i < 4096; i += 256) {
    int r = i >> 6, c = i & 63;                       // i = l2*64 + l1
    pinvT[(size_t)bh * 4096 + i] = (_Float16)V[c * 64 + r];
  }
}

// ---------------------------------------------------------------------------
// K5: new_value = softmax(ql @ k^T, axis=S) @ v  -- flash-style online softmax
// over S in 64-wide chunks; logits and P@V both WMMA; output stored transposed
// (nvT[d][l]) for async B staging downstream.
// ---------------------------------------------------------------------------
__global__ void __launch_bounds__(128)
newvalue_kernel(const _Float16* __restrict__ ql, const _Float16* __restrict__ k,
                const _Float16* __restrict__ vT, _Float16* __restrict__ nvT) {
  __shared__ _Float16 Aql[64 * LP], Bk[64 * LP], Bv[64 * LP], Pm[64 * LP];
  __shared__ float Lg[64 * 64];
  __shared__ float mrow[64], srow[64], scl[64];
  const int bh = blockIdx.x, tid = threadIdx.x, wave = tid >> 5, lane = tid & 31;
  async_tile8(Aql, LP, ql + (size_t)bh * 4096, 64, 64, tid);
  if (tid < 64) { mrow[tid] = -3.0e38f; srow[tid] = 0.f; }
  v8f acc[4] = {};
  async_wait();
  __syncthreads();
  for (int s0 = 0; s0 < SS; s0 += 64) {
    async_tile8(Bk, LP, k + ((size_t)bh * SS + s0) * DD, DD, 64, tid);   // k rows
    async_tile8(Bv, LP, vT + (size_t)bh * DD * SS + s0, SS, 64, tid);    // v^T rows
    async_wait();
    __syncthreads();
    v8f lacc[4] = {};
    wmma_accum_g<64>(lacc, Aql, LP, Bk, LP, wave, lane);  // logits = ql @ k^T
    store_acc_lds(lacc, Lg, 64, wave, lane);
    __syncthreads();
    if (tid < 64) {
      float mx = mrow[tid];
      for (int c = 0; c < 64; ++c) mx = fmaxf(mx, Lg[tid * 64 + c]);
      float sc = __expf(mrow[tid] - mx);
      float ss = srow[tid] * sc;
      for (int c = 0; c < 64; ++c) {
        float p = __expf(Lg[tid * 64 + c] - mx);
        Pm[tid * LP + c] = (_Float16)p;
        ss += p;
      }
      mrow[tid] = mx; srow[tid] = ss; scl[tid] = sc;
    }
    __syncthreads();
#pragma unroll
    for (int j = 0; j < 4; ++j)
#pragma unroll
      for (int i = 0; i < 8; ++i) {
        int rowi = wave * 16 + i + ((lane & 16) ? 8 : 0);
        acc[j][i] *= scl[rowi];  // rescale running accumulator
      }
    wmma_accum_g<64>(acc, Pm, LP, Bv, LP, wave, lane);  // acc += P @ v
    __syncthreads();
  }
  store_acc_lds(acc, Lg, 64, wave, lane);
  __syncthreads();
  for (int idx = tid; idx < 4096; idx += 128) {
    int d = idx >> 6, l = idx & 63;
    nvT[(size_t)bh * 4096 + idx] = (_Float16)(Lg[l * 64 + d] / srow[l]);
  }
}

// ---------------------------------------------------------------------------
// K6: fused per-64-row tile:
//   ctx = softmax(q@kl^T) @ pinv @ new_value  (three K=64 WMMAs)
//   + depthwise conv as a BANDED WMMA (64x96 band of conv weights times the
//     96-row v^T window), accumulated into the SAME C fragments.
// grid = (B*H, S/64).
// ---------------------------------------------------------------------------
__global__ void __launch_bounds__(128)
ctx_kernel(const _Float16* __restrict__ q, const _Float16* __restrict__ kl,
           const _Float16* __restrict__ pinvT, const _Float16* __restrict__ nvT,
           const _Float16* __restrict__ vT, const float* __restrict__ conv_w,
           float* __restrict__ out) {
  __shared__ _Float16 AqBand[64 * LPW];   // Aq (stride LP) then conv band (LPW)
  __shared__ _Float16 Bt[64 * LPW];       // kl / pinvT / nvT (LP) then v win (LPW)
  __shared__ _Float16 P1[64 * LP];
  __shared__ float Lg[64 * 64];
  __shared__ float wgt[KSZ];
  const int bh = blockIdx.x, tid = threadIdx.x, wave = tid >> 5, lane = tid & 31;
  const int b = bh >> 4, h = bh & 15;
  const int s0 = blockIdx.y * 64;
  if (tid < KSZ) wgt[tid] = conv_w[h * KSZ + tid];
  async_tile8(AqBand, LP, q + ((size_t)bh * SS + s0) * DD, DD, 64, tid);
  async_tile8(Bt, LP, kl + (size_t)bh * 4096, 64, 64, tid);
  async_wait();
  __syncthreads();
  {
    v8f acc[4] = {};
    wmma_accum_g<64>(acc, AqBand, LP, Bt, LP, wave, lane);  // logits = q @ kl^T
    store_acc_lds(acc, Lg, 64, wave, lane);
  }
  __syncthreads();
  if (tid < 64) {  // kernel_1 softmax over L=64
    float mx = -3.0e38f;
    for (int c = 0; c < 64; ++c) mx = fmaxf(mx, Lg[tid * 64 + c]);
    float s = 0.f;
    for (int c = 0; c < 64; ++c) {
      float e = __expf(Lg[tid * 64 + c] - mx);
      Lg[tid * 64 + c] = e; s += e;
    }
    float inv = 1.f / s;
    for (int c = 0; c < 64; ++c) P1[tid * LP + c] = (_Float16)(Lg[tid * 64 + c] * inv);
  }
  async_tile8(Bt, LP, pinvT + (size_t)bh * 4096, 64, 64, tid);  // B-T of pinv
  async_wait();
  __syncthreads();
  {
    v8f acc[4] = {};
    wmma_accum_g<64>(acc, P1, LP, Bt, LP, wave, lane);  // probs = k1 @ pinv
    store_acc_lds(acc, Lg, 64, wave, lane);
  }
  __syncthreads();
  for (int idx = tid; idx < 4096; idx += 128) {
    int r = idx >> 6, c = idx & 63;
    P1[r * LP + c] = (_Float16)Lg[idx];
  }
  async_tile8(Bt, LP, nvT + (size_t)bh * 4096, 64, 64, tid);    // B-T of nv
  async_wait();
  __syncthreads();
  v8f acc[4] = {};
  wmma_accum_g<64>(acc, P1, LP, Bt, LP, wave, lane);  // ctx = probs @ nv
  __syncthreads();  // all reads of Bt/P1 done; refill for conv
  // Conv band: AqBand[s][j] = wgt[j-s+16] on band, else 0 (j in [0,96)).
  for (int idx = tid; idx < 64 * 128; idx += 128) {
    int s = idx >> 7, j = idx & 127;
    if (j < 96) {
      int kk = j - s + KSZ / 2;
      AqBand[s * LPW + j] = (kk >= 0 && kk < KSZ) ? (_Float16)wgt[kk] : (_Float16)0.f;
    }
  }
  // v^T window rows d, cols j -> seq s0-16+j  (96 cols).
  if (s0 >= 16 && s0 + 80 <= SS) {
    async_tile12(Bt, LPW, vT + (size_t)bh * DD * SS + (s0 - 16), SS, tid);
    async_wait();
  } else {
    for (int idx = tid; idx < 64 * 128; idx += 128) {
      int d = idx >> 7, j = idx & 127;
      if (j < 96) {
        int si = s0 - 16 + j;
        Bt[d * LPW + j] = (si >= 0 && si < SS)
                              ? vT[((size_t)bh * DD + d) * SS + si] : (_Float16)0.f;
      }
    }
  }
  __syncthreads();
  wmma_accum_g<96>(acc, AqBand, LPW, Bt, LPW, wave, lane);  // += conv (banded)
  store_acc_lds(acc, Lg, 64, wave, lane);
  __syncthreads();
  for (int idx = tid; idx < 4096; idx += 128) {
    int r = idx >> 6, d = idx & 63;
    out[((size_t)b * SS + s0 + r) * (HH * DD) + h * DD + d] = Lg[idx];
  }
}

// ---------------------------------------------------------------------------
// Host launch
// ---------------------------------------------------------------------------
extern "C" void kernel_launch(void* const* d_in, const int* in_sizes, int n_in,
                              void* d_out, int out_size, void* d_ws, size_t ws_size,
                              hipStream_t stream) {
  (void)in_sizes; (void)n_in; (void)out_size; (void)ws_size;
  const float* hidden = (const float*)d_in[0];
  const float* Wq     = (const float*)d_in[1];
  const float* bq     = (const float*)d_in[2];
  const float* Wk     = (const float*)d_in[3];
  const float* bk     = (const float*)d_in[4];
  const float* Wv     = (const float*)d_in[5];
  const float* bv     = (const float*)d_in[6];
  const float* conv_w = (const float*)d_in[7];
  float* out = (float*)d_out;

  char* w = (char*)d_ws;
  auto take = [&](size_t bytes) -> char* {
    char* p = w;
    w += (bytes + 255) & ~(size_t)255;
    return p;
  };
  const size_t wbytes   = (size_t)EE * EE * sizeof(_Float16);
  const size_t qkvbytes = (size_t)BB * HH * SS * DD * sizeof(_Float16);
  const size_t lmbytes  = (size_t)BB * HH * LL * DD * sizeof(_Float16);

  _Float16* X16    = (_Float16*)take((size_t)BB * SS * EE * sizeof(_Float16));
  _Float16* WTq    = (_Float16*)take(wbytes);
  _Float16* WTk    = (_Float16*)take(wbytes);
  _Float16* WTv    = (_Float16*)take(wbytes);
  _Float16* q16    = (_Float16*)take(qkvbytes);
  _Float16* k16    = (_Float16*)take(qkvbytes);
  _Float16* v16    = (_Float16*)take(qkvbytes);
  _Float16* vT16   = (_Float16*)take(qkvbytes);
  _Float16* ql16   = (_Float16*)take(lmbytes);
  _Float16* kl16   = (_Float16*)take(lmbytes);
  float*    k2     = (float*)take((size_t)BB * HH * LL * LL * sizeof(float));
  _Float16* pinvT  = (_Float16*)take((size_t)BB * HH * LL * LL * sizeof(_Float16));
  _Float16* nvT    = (_Float16*)take((size_t)BB * HH * LL * DD * sizeof(_Float16));
  unsigned* gmax   = (unsigned*)take(sizeof(unsigned));

  convert_hidden_kernel<<<(BB * SS * EE) / 1024, 256, 0, stream>>>(hidden, X16);
  dim3 gtw(EE / 64, EE / 64);
  transposeW_kernel<<<gtw, 128, 0, stream>>>(Wq, WTq);
  transposeW_kernel<<<gtw, 128, 0, stream>>>(Wk, WTk);
  transposeW_kernel<<<gtw, 128, 0, stream>>>(Wv, WTv);

  const float scale = 0.35355339059327373f;  // 64^-0.25
  dim3 ggemm(BB * SS / 128, HH);
  qkv_gemm_kernel<<<ggemm, 128, 0, stream>>>(X16, WTq, bq, scale, q16);
  qkv_gemm_kernel<<<ggemm, 128, 0, stream>>>(X16, WTk, bk, scale, k16);
  qkv_gemm_kernel<<<ggemm, 128, 0, stream>>>(X16, WTv, bv, 1.0f, v16);

  transposeV_kernel<<<dim3(BB * HH, SS / 64), 128, 0, stream>>>(v16, vT16);
  landmark_kernel<<<BB * HH * LL, 64, 0, stream>>>(q16, k16, ql16, kl16);

  zero_u32_kernel<<<1, 1, 0, stream>>>(gmax);
  kernel2_kernel<<<BB * HH, 128, 0, stream>>>(ql16, kl16, k2, gmax);
  pinv_kernel<<<BB * HH, 256, 0, stream>>>(k2, gmax, pinvT);

  newvalue_kernel<<<BB * HH, 128, 0, stream>>>(ql16, k16, vT16, nvT);

  ctx_kernel<<<dim3(BB * HH, SS / 64), 128, 0, stream>>>(
      q16, kl16, pinvT, nvT, vT16, conv_w, out);
}
